// CausalSelfAttention_1846835937317
// MI455X (gfx1250) — matrix operand order repaired and checked by reading, not compiled, not run on hardware
//
#include <hip/hip_runtime.h>
#include <hip/hip_bf16.h>

// ---------------------------------------------------------------------------
// Causal self-attention for MI455X (gfx1250), bf16 WMMA pipeline.
//   x[2,2048,1024] f32, W_qkv[3072,1024] f32, W_proj[1024,1024] f32 -> out f32
// Compute-bound (~1000 FLOP/byte) => everything through v_wmma_f32_16x16x32_bf16.
// GEMMs: 32x64 wave tiles, ping-pong double-buffered K loop (no copy movs).
// Attention: flash-style online softmax. V is stored TRANSPOSED [B,H,Dh,T] by
// the QKV GEMM so P*V B-tiles are contiguous; V tiles are streamed into
// double-buffered LDS by the Tensor Data Mover (TDM) one key-tile ahead,
// synchronized with s_wait_tensorcnt(1) -> genuine compute/DMA overlap.
// ---------------------------------------------------------------------------

typedef __bf16 bf16_t;
typedef bf16_t v16bf __attribute__((ext_vector_type(16)));
typedef bf16_t v8bf  __attribute__((ext_vector_type(8)));
typedef bf16_t v4bf  __attribute__((ext_vector_type(4)));
typedef bf16_t v2bf  __attribute__((ext_vector_type(2)));
typedef float  v8f   __attribute__((ext_vector_type(8)));
typedef int    v4i_t __attribute__((ext_vector_type(4)));
typedef unsigned int v4u_t __attribute__((ext_vector_type(4)));
typedef int    v8i_t __attribute__((ext_vector_type(8)));

#define BN 2
#define TT 2048
#define CC 1024
#define NH 16
#define DH 64
#define BT (BN * TT)          // 4096 rows of x
#define NQKV (3 * CC)         // 3072

// --------------------------- feature probes --------------------------------
#if __has_builtin(__builtin_amdgcn_tensor_load_to_lds) && \
    __has_builtin(__builtin_amdgcn_s_wait_tensorcnt)
#define HAVE_TDM 1
// Tensor DMA Descriptor (ISA cdna5 ch.8): 2D tile load, data_size = 2 bytes.
//   dim0 = contiguous elements, dim1 = rows.
// Round-5 probe: builtin takes 6 args (g0 v4u, g1 v8i, g2 v4i, g3 v4i,
// extra v8i, i32 cpol).
__device__ __forceinline__ void tdm_load_tile_2d(const void* gsrc, void* ldst,
                                                 unsigned tile_d0, unsigned tile_d1,
                                                 unsigned tensor_d0, unsigned tensor_d1,
                                                 unsigned stride_d0 /*elements*/) {
  unsigned long long ga = (unsigned long long)gsrc;
  unsigned la = (unsigned)(unsigned long long)ldst;   // LDS byte address
  v4u_t g0;
  g0[0] = 1u;                                         // count=1 (valid user D#)
  g0[1] = la;                                         // lds_addr [63:32]
  g0[2] = (unsigned)(ga & 0xffffffffu);               // global_addr [95:64]
  g0[3] = (unsigned)((ga >> 32) & 0x01ffffffu)        // global_addr [120:96]
        | (2u << 30);                                 // type = 2 ("image")
  v8i_t g1;
  g1[0] = (int)(1u << 16);                            // wg_mask=0, data_size=1 (2B)
  g1[1] = (int)((tensor_d0 & 0xffffu) << 16);         // tensor_dim0[15:0] @ bit48
  g1[2] = (int)(((tensor_d0 >> 16) & 0xffffu) |
                ((tensor_d1 & 0xffffu) << 16));       // dim0 hi | dim1 lo
  g1[3] = (int)(((tensor_d1 >> 16) & 0xffffu) |
                ((tile_d0 & 0xffffu) << 16));         // dim1 hi | tile_dim0
  g1[4] = (int)(tile_d1 & 0xffffu);                   // tile_dim1 (tile_dim2=0)
  g1[5] = (int)stride_d0;                             // tensor_dim0_stride[31:0]
  g1[6] = 0;                                          // stride hi, dim1_stride lo
  g1[7] = 0;
  v4i_t z4 = {0, 0, 0, 0};
  v8i_t z8 = {0, 0, 0, 0, 0, 0, 0, 0};
  __builtin_amdgcn_tensor_load_to_lds(g0, g1, z4, z4, z8, 0);
}
#else
#define HAVE_TDM 0
#endif

#if __has_builtin(__builtin_amdgcn_global_load_async_to_lds_b128) && \
    __has_builtin(__builtin_amdgcn_s_wait_asynccnt)
#define HAVE_ASYNC 1
// Round-2 probe: param 0 is `int __vector(4) __device__ *` (global AS).
typedef __attribute__((address_space(1))) v4i_t as1_v4i;
typedef __attribute__((address_space(3))) v4i_t as3_v4i;
#else
#define HAVE_ASYNC 0
#endif

__device__ __forceinline__ bf16_t f32_to_bf16(float f) {
  union { float f; unsigned int u; } in; in.f = f;
  unsigned int u = in.u;
  u += 0x7fffu + ((u >> 16) & 1u);           // round-to-nearest-even
  union { unsigned short s; bf16_t b; } out; out.s = (unsigned short)(u >> 16);
  return out.b;
}

__device__ __forceinline__ v2bf f32x2_to_bf16x2(float a, float b) {
#if __has_builtin(__builtin_amdgcn_cvt_pk_bf16_f32)
  auto pk = __builtin_amdgcn_cvt_pk_bf16_f32(a, b);
  return __builtin_bit_cast(v2bf, pk);
#else
  v2bf r; r[0] = f32_to_bf16(a); r[1] = f32_to_bf16(b); return r;
#endif
}

__device__ __forceinline__ v8f wmma_bf16(v16bf a, v16bf b, v8f c) {
  // D = A(16x32) * B(32x16) + C(16x16 f32)
  return __builtin_amdgcn_wmma_f32_16x16x32_bf16(false, a, false, b,
                                                 (short)0, c, false, false);
}

// A-matrix 16x32 bf16 (ISA 7.12.2): lane L<16 -> row L, elems 0..7 = K 0..7,
// elems 8..15 = K 16..23; lane L>=16 -> row L-16, K 8..15 / 24..31.
__device__ __forceinline__ v16bf load_A16x32(const bf16_t* base, int ld, int lane) {
  int row  = lane & 15;
  int kofs = (lane >> 4) << 3;               // 0 or 8
  const v8bf* p0 = (const v8bf*)(base + (size_t)row * ld + kofs);
  const v8bf* p1 = (const v8bf*)(base + (size_t)row * ld + kofs + 16);
  v8bf lo = *p0, hi = *p1;
  return __builtin_shufflevector(lo, hi, 0,1,2,3,4,5,6,7,8,9,10,11,12,13,14,15);
}

// B-matrix 32x16 bf16: lane holds column n = lane&15; lanes 0-15 K 0..7/16..23,
// lanes 16-31 K 8..15/24..31.  Element (k,n) lives at base + n*ld + k.
__device__ __forceinline__ v16bf load_B32x16(const bf16_t* base, int ld, int lane) {
  int n  = lane & 15;
  int k2 = (lane >> 4) << 3;                 // 0 or 8
  const v8bf* p0 = (const v8bf*)(base + (size_t)n * ld + k2);
  const v8bf* p1 = (const v8bf*)(base + (size_t)n * ld + k2 + 16);
  v8bf lo = *p0, hi = *p1;
  return __builtin_shufflevector(lo, hi, 0,1,2,3,4,5,6,7,8,9,10,11,12,13,14,15);
}

// ---------------------------------------------------------------------------
// fp32 -> bf16 conversion (vectorized 4-wide)
// ---------------------------------------------------------------------------
__global__ void __launch_bounds__(256)
cvt_f32_bf16_kernel(const float* __restrict__ s, bf16_t* __restrict__ d, int n4) {
  int i = blockIdx.x * blockDim.x + threadIdx.x;
  if (i < n4) {
    float4 f = ((const float4*)s)[i];
    v2bf lo = f32x2_to_bf16x2(f.x, f.y);
    v2bf hi = f32x2_to_bf16x2(f.z, f.w);
    v4bf o = __builtin_shufflevector(lo, hi, 0, 1, 2, 3);
    *((v4bf*)(d + (size_t)i * 4)) = o;
  }
}

// ---------------------------------------------------------------------------
// Ping-pong pipelined 32(M) x 64(N) wave-tile GEMM core, K-step 32.
// ---------------------------------------------------------------------------
struct Tiles {
  v16bf a0, a1;        // rows m..m+15, m+16..m+31
  v16bf b0, b1, b2, b3;
};

__device__ __forceinline__ Tiles load_tiles(const bf16_t* __restrict__ X,
                                            const bf16_t* __restrict__ W,
                                            int m_base, int nb, int k, int lane) {
  Tiles t;
  t.a0 = load_A16x32(X + (size_t)m_base * CC + k, CC, lane);
  t.a1 = load_A16x32(X + (size_t)(m_base + 16) * CC + k, CC, lane);
  t.b0 = load_B32x16(W + (size_t)(nb +  0) * CC + k, CC, lane);
  t.b1 = load_B32x16(W + (size_t)(nb + 16) * CC + k, CC, lane);
  t.b2 = load_B32x16(W + (size_t)(nb + 32) * CC + k, CC, lane);
  t.b3 = load_B32x16(W + (size_t)(nb + 48) * CC + k, CC, lane);
  return t;
}

__device__ __forceinline__ void mm_step(const Tiles& t, v8f acc[8]) {
  acc[0] = wmma_bf16(t.a0, t.b0, acc[0]);
  acc[1] = wmma_bf16(t.a0, t.b1, acc[1]);
  acc[2] = wmma_bf16(t.a0, t.b2, acc[2]);
  acc[3] = wmma_bf16(t.a0, t.b3, acc[3]);
  acc[4] = wmma_bf16(t.a1, t.b0, acc[4]);
  acc[5] = wmma_bf16(t.a1, t.b1, acc[5]);
  acc[6] = wmma_bf16(t.a1, t.b2, acc[6]);
  acc[7] = wmma_bf16(t.a1, t.b3, acc[7]);
}

__device__ __forceinline__ void gemm_core(const bf16_t* __restrict__ X,
                                          const bf16_t* __restrict__ W,
                                          int m_base, int nb, int lane,
                                          v8f acc[8]) {
  #pragma unroll
  for (int j = 0; j < 8; ++j)
    #pragma unroll
    for (int i = 0; i < 8; ++i) acc[j][i] = 0.f;

  Tiles ta = load_tiles(X, W, m_base, nb, 0, lane);
  Tiles tb = load_tiles(X, W, m_base, nb, 32, lane);
  #pragma unroll 1
  for (int k = 64; k < CC; k += 64) {
    mm_step(ta, acc);
    ta = load_tiles(X, W, m_base, nb, k, lane);
    mm_step(tb, acc);
    tb = load_tiles(X, W, m_base, nb, k + 32, lane);
  }
  mm_step(ta, acc);
  mm_step(tb, acc);
}

// ---------------------------------------------------------------------------
// QKV GEMM: qkv[m][o] = sum_c x[m][c] * W_qkv[o][c]
// Q,K scattered to [B,H,T,DH]; V scattered TRANSPOSED to [B,H,DH,T] so the
// flash kernel's P*V B-tiles are key-contiguous.
// ---------------------------------------------------------------------------
__global__ void __launch_bounds__(256, 2)
gemm_qkv_kernel(const bf16_t* __restrict__ X,
                const bf16_t* __restrict__ W,
                bf16_t* __restrict__ Qp,
                bf16_t* __restrict__ Kp,
                bf16_t* __restrict__ Vp) {
  int lane = threadIdx.x & 31;
  int wave = threadIdx.x >> 5;
  int m_base = blockIdx.x * 32;
  int nb     = blockIdx.y * 512 + wave * 64;

  v8f acc[8];
  gemm_core(X, W, m_base, nb, lane, acc);

  int n   = lane & 15;
  int hi8 = (lane >> 4) << 3;
  #pragma unroll
  for (int j = 0; j < 4; ++j) {
    int o     = nb + j * 16 + n;
    int which = o >> 10;            // 0:q 1:k 2:v
    int rem   = o & 1023;
    int h     = rem >> 6;
    int d     = rem & 63;
    #pragma unroll
    for (int mi = 0; mi < 2; ++mi) {
      #pragma unroll
      for (int r = 0; r < 8; ++r) {
        int m = m_base + mi * 16 + r + hi8;
        int b = m >> 11, t = m & (TT - 1);
        bf16_t val = f32_to_bf16(acc[mi * 4 + j][r]);
        if (which == 0)
          Qp[(((size_t)b * NH + h) * TT + t) * DH + d] = val;
        else if (which == 1)
          Kp[(((size_t)b * NH + h) * TT + t) * DH + d] = val;
        else
          Vp[(((size_t)b * NH + h) * DH + d) * TT + t] = val;   // transposed
      }
    }
  }
}

// ---------------------------------------------------------------------------
// Flash attention: one wave per (b, h, 16-query tile). Online softmax in the
// f32 C-layout; P transposed to A-layout through per-wave LDS.  V^T tiles
// (64 d-rows x 32 keys) stream into double-buffered LDS via TDM, one key-tile
// ahead of compute; P*V B-tiles are then contiguous 16B LDS loads.
// ---------------------------------------------------------------------------
__global__ void __launch_bounds__(256, 2)
flash_attn_kernel(const bf16_t* __restrict__ Qg,
                  const bf16_t* __restrict__ Kg,
                  const bf16_t* __restrict__ Vtg,   // [B,H,DH,T]
                  bf16_t* __restrict__ Y) {
  __shared__ __align__(16) bf16_t ldsVt[8][2][DH * 32];  // 64 KB (2 x 4KB/wave)
  __shared__ __align__(16) bf16_t ldsP[8][16 * 32];      //  8 KB

  int lane = threadIdx.x & 31;
  int wave = threadIdx.x >> 5;
  int idx  = blockIdx.x * 8 + wave;     // 4096 waves total = 2*16*128
  int qt   = idx & 127;
  int h    = (idx >> 7) & 15;
  int b    = idx >> 11;

  const bf16_t* Qb  = Qg  + ((size_t)b * NH + h) * TT * DH;
  const bf16_t* Kb  = Kg  + ((size_t)b * NH + h) * TT * DH;
  const bf16_t* Vtb = Vtg + ((size_t)b * NH + h) * DH * TT;  // rows d, ld = TT

  int n   = lane & 15;
  int hi8 = (lane >> 4) << 3;

  // Q tile (16 x 64) in A-layout, loaded once.
  v16bf aq0 = load_A16x32(Qb + (size_t)qt * 16 * DH, DH, lane);       // d 0..31
  v16bf aq1 = load_A16x32(Qb + (size_t)qt * 16 * DH + 32, DH, lane);  // d 32..63

  float mrow[8], lrow[8];
  v8f acc[4];
  #pragma unroll
  for (int r = 0; r < 8; ++r) { mrow[r] = -1e30f; lrow[r] = 0.f; }
  #pragma unroll
  for (int j = 0; j < 4; ++j)
    #pragma unroll
    for (int i = 0; i < 8; ++i) acc[j][i] = 0.f;

  int kt_end = (qt * 16 + 15) >> 5;     // inclusive; causal limit

#if HAVE_TDM
  // Prime the pipeline: TDM streams V^T tile 0 (64 rows x 32 keys).
  tdm_load_tile_2d(Vtb, &ldsVt[wave][0][0],
                   /*tile_d0=*/32, /*tile_d1=*/DH,
                   /*tensor_d0=*/TT, /*tensor_d1=*/DH, /*stride_d0=*/TT);
#endif

  #pragma unroll 1
  for (int kt = 0; kt <= kt_end; ++kt) {
    int key0 = kt << 5;

#if HAVE_TDM
    // Issue DMA for the NEXT key-tile while we compute on the current one.
    if (kt < kt_end)
      tdm_load_tile_2d(Vtb + (size_t)(key0 + 32), &ldsVt[wave][(kt + 1) & 1][0],
                       32, DH, TT, DH, TT);
#elif HAVE_ASYNC
    {
      // Each lane stages two 32-key rows (d = 2*lane, 2*lane+1).
      const bf16_t* srcb = Vtb + (size_t)(lane * 2) * TT + key0;
      bf16_t* dstb = &ldsVt[wave][kt & 1][0] + lane * 64;
      as1_v4i* g0p = (as1_v4i*)(unsigned long long)srcb;
      as1_v4i* g1p = (as1_v4i*)(unsigned long long)(srcb + TT);
      as3_v4i* l0p = (as3_v4i*)(unsigned)(unsigned long long)dstb;
      #pragma unroll
      for (int i = 0; i < 4; ++i) {
        __builtin_amdgcn_global_load_async_to_lds_b128(g0p + i, l0p + i, 0, 0);
        __builtin_amdgcn_global_load_async_to_lds_b128(g1p + i, l0p + 4 + i, 0, 0);
      }
    }
#else
    {
      const v8bf* s0 = (const v8bf*)(Vtb + (size_t)(lane * 2) * TT + key0);
      const v8bf* s1 = (const v8bf*)(Vtb + (size_t)(lane * 2 + 1) * TT + key0);
      v8bf* dd = (v8bf*)(&ldsVt[wave][kt & 1][0] + lane * 64);
      #pragma unroll
      for (int i = 0; i < 4; ++i) { dd[i] = s0[i]; dd[4 + i] = s1[i]; }
    }
#endif
    // Prefetch next K tile (global_prefetch_b8).
    if (kt < kt_end)
      __builtin_prefetch(Kb + (size_t)(key0 + 32 + lane) * DH, 0, 1);

    // S = Q * K^T (16 queries x 32 keys): batch the 4 B-tile loads, then WMMA.
    v8f s0, s1;
    #pragma unroll
    for (int i = 0; i < 8; ++i) { s0[i] = 0.f; s1[i] = 0.f; }
    {
      v16bf bk0 = load_B32x16(Kb + (size_t)key0 * DH + 0,  DH, lane);
      v16bf bk1 = load_B32x16(Kb + (size_t)key0 * DH + 32, DH, lane);
      v16bf bk2 = load_B32x16(Kb + (size_t)(key0 + 16) * DH + 0,  DH, lane);
      v16bf bk3 = load_B32x16(Kb + (size_t)(key0 + 16) * DH + 32, DH, lane);
      s0 = wmma_bf16(aq0, bk0, s0);
      s0 = wmma_bf16(aq1, bk1, s0);
      s1 = wmma_bf16(aq0, bk2, s1);
      s1 = wmma_bf16(aq1, bk3, s1);
    }

    // Mask + online softmax (row r in VGPR r; half-wave selects rows 0-7/8-15).
    const float sc = 0.125f;  // 1/sqrt(64)
    #pragma unroll
    for (int r = 0; r < 8; ++r) {
      int qrow = qt * 16 + r + hi8;
      float x0 = s0[r] * sc; if (key0 + n      > qrow) x0 = -1e30f;
      float x1 = s1[r] * sc; if (key0 + 16 + n > qrow) x1 = -1e30f;
      float mx = fmaxf(x0, x1);
      #pragma unroll
      for (int off = 1; off < 16; off <<= 1)
        mx = fmaxf(mx, __shfl_xor(mx, off, 32));
      float mnew    = fmaxf(mrow[r], mx);
      float rescale = __expf(mrow[r] - mnew);
      mrow[r] = mnew;
      float p0 = __expf(x0 - mnew);
      float p1 = __expf(x1 - mnew);
      float rs = p0 + p1;
      #pragma unroll
      for (int off = 1; off < 16; off <<= 1)
        rs += __shfl_xor(rs, off, 32);
      lrow[r] = lrow[r] * rescale + rs;
      #pragma unroll
      for (int j = 0; j < 4; ++j) acc[j][r] *= rescale;
      // C-layout -> row-major P tile [16][32] in LDS.
      ldsP[wave][(r + hi8) * 32 + n]      = f32_to_bf16(p0);
      ldsP[wave][(r + hi8) * 32 + 16 + n] = f32_to_bf16(p1);
    }
    __builtin_amdgcn_wave_barrier();

#if HAVE_TDM
    // Current tile done when <=1 TDM op outstanding (next tile still streams).
    if (kt < kt_end) __builtin_amdgcn_s_wait_tensorcnt(1);
    else             __builtin_amdgcn_s_wait_tensorcnt(0);
#elif HAVE_ASYNC
    __builtin_amdgcn_s_wait_asynccnt(0);
#endif

    // P back as A-matrix 16x32; V^T B-tiles are contiguous LDS rows (ld=32).
    v16bf ap = load_A16x32(&ldsP[wave][0], 32, lane);
    const bf16_t* vb0 = &ldsVt[wave][kt & 1][0];
    #pragma unroll
    for (int j = 0; j < 4; ++j) {
      v16bf bv = load_B32x16(vb0 + (size_t)j * 16 * 32, 32, lane);
      acc[j] = wmma_bf16(ap, bv, acc[j]);
    }
  }

  // Epilogue: O / l, store bf16 into y[B,T,C] (head-interleaved) for proj GEMM.
  bf16_t* yb = Y + (size_t)b * TT * CC + h * DH;
  #pragma unroll
  for (int j = 0; j < 4; ++j)
    #pragma unroll
    for (int r = 0; r < 8; ++r) {
      int qrow = qt * 16 + r + hi8;
      yb[(size_t)qrow * CC + j * 16 + n] = f32_to_bf16(acc[j][r] / lrow[r]);
    }
}

// ---------------------------------------------------------------------------
// Output projection: out[m][o] = sum_c y[m][c] * W_proj[o][c], f32 out.
// ---------------------------------------------------------------------------
__global__ void __launch_bounds__(256, 2)
gemm_proj_kernel(const bf16_t* __restrict__ Ym,
                 const bf16_t* __restrict__ W,
                 float* __restrict__ Out) {
  int lane = threadIdx.x & 31;
  int wave = threadIdx.x >> 5;
  int m_base = blockIdx.x * 32;
  int nb     = blockIdx.y * 512 + wave * 64;

  v8f acc[8];
  gemm_core(Ym, W, m_base, nb, lane, acc);

  int n   = lane & 15;
  int hi8 = (lane >> 4) << 3;
  #pragma unroll
  for (int j = 0; j < 4; ++j) {
    int o = nb + j * 16 + n;
    #pragma unroll
    for (int mi = 0; mi < 2; ++mi) {
      #pragma unroll
      for (int r = 0; r < 8; ++r) {
        int m = m_base + mi * 16 + r + hi8;
        Out[(size_t)m * CC + o] = acc[mi * 4 + j][r];
      }
    }
  }
}

// ---------------------------------------------------------------------------
extern "C" void kernel_launch(void* const* d_in, const int* in_sizes, int n_in,
                              void* d_out, int out_size, void* d_ws, size_t ws_size,
                              hipStream_t stream) {
  const float* x      = (const float*)d_in[0];   // [2,2048,1024]
  const float* W_qkv  = (const float*)d_in[1];   // [3072,1024]
  const float* W_proj = (const float*)d_in[2];   // [1024,1024]
  float* out = (float*)d_out;                    // [2,2048,1024]

  // Workspace layout (bf16), 48 MB total.
  char* ws = (char*)d_ws;
  bf16_t* xb     = (bf16_t*)ws;                          ws += (size_t)BT * CC * 2;
  bf16_t* wqkvb  = (bf16_t*)ws;                          ws += (size_t)NQKV * CC * 2;
  bf16_t* wprojb = (bf16_t*)ws;                          ws += (size_t)CC * CC * 2;
  bf16_t* qb     = (bf16_t*)ws;                          ws += (size_t)BT * CC * 2;
  bf16_t* kb     = (bf16_t*)ws;                          ws += (size_t)BT * CC * 2;
  bf16_t* vtb    = (bf16_t*)ws;                          ws += (size_t)BT * CC * 2;
  bf16_t* yb     = (bf16_t*)ws;                          ws += (size_t)BT * CC * 2;
  (void)ws_size; (void)in_sizes; (void)n_in; (void)out_size;

  // 1) fp32 -> bf16 conversions.
  {
    int n4 = (BT * CC) / 4;
    cvt_f32_bf16_kernel<<<(n4 + 255) / 256, 256, 0, stream>>>(x, xb, n4);
    n4 = (NQKV * CC) / 4;
    cvt_f32_bf16_kernel<<<(n4 + 255) / 256, 256, 0, stream>>>(W_qkv, wqkvb, n4);
    n4 = (CC * CC) / 4;
    cvt_f32_bf16_kernel<<<(n4 + 255) / 256, 256, 0, stream>>>(W_proj, wprojb, n4);
  }

  // 2) QKV projection: [4096,1024] x [1024,3072] via bf16 WMMA (32x64/wave).
  {
    dim3 grid(BT / 32, NQKV / 512);     // 128 x 6
    gemm_qkv_kernel<<<grid, 256, 0, stream>>>(xb, wqkvb, qb, kb, vtb);
  }

  // 3) Flash attention: 4096 waves (2 * 16 heads * 128 q-tiles), 8 waves/block.
  {
    flash_attn_kernel<<<512, 256, 0, stream>>>(qb, kb, vtb, yb);
  }

  // 4) Output projection: [4096,1024] x [1024,1024] -> f32 out.
  {
    dim3 grid(BT / 32, CC / 512);       // 128 x 2
    gemm_proj_kernel<<<grid, 256, 0, stream>>>(yb, wprojb, out);
  }
}
